// HypergraphGNN_7576322310517
// MI455X (gfx1250) — matrix operand order repaired and checked by reading, not compiled, not run on hardware
//
#include <hip/hip_runtime.h>
#include <hip/hip_bf16.h>
#include <math.h>

// ---------------- CDNA5 WMMA types ----------------
typedef __bf16 v16bf __attribute__((ext_vector_type(16)));
typedef __bf16 v8bf  __attribute__((ext_vector_type(8)));
typedef float  v8f   __attribute__((ext_vector_type(8)));

#define Bc    4
#define Pc    8
#define Nc    2000
#define Ec    2000
#define INc   16
#define HIDc  128
#define NLc   3
#define MULTc 4
#define NPAD  2048   // N and E padded to tile multiple (also padded K-extent)

// ---------------- zero workspace (grid-stride, 16B stores) ----------------
__global__ __launch_bounds__(256) void zero_ws(uint4* __restrict__ p, size_t count16) {
    size_t i = (size_t)blockIdx.x * blockDim.x + threadIdx.x;
    size_t stride = (size_t)gridDim.x * blockDim.x;
    uint4 z; z.x = 0; z.y = 0; z.z = 0; z.w = 0;
    for (; i < count16; i += stride) p[i] = z;
}

// ---------------- f32 -> bf16 convert (contiguous) ----------------
__global__ __launch_bounds__(256) void cvt_f32_bf16(const float* __restrict__ in,
                                                    __bf16* __restrict__ out,
                                                    long count) {
    long i = (long)blockIdx.x * blockDim.x + threadIdx.x;
    long stride = (long)gridDim.x * blockDim.x;
    for (; i < count; i += stride) out[i] = (__bf16)in[i];
}

// ---------------- f32 -> bf16 convert into padded layout ----------------
__global__ __launch_bounds__(256) void cvt_pad_bf16(const float* __restrict__ in,
                                                    __bf16* __restrict__ out,
                                                    int cols, int ldo,
                                                    size_t inBatch, size_t outBatch) {
    int b = blockIdx.z;
    int r = blockIdx.y;
    int c = blockIdx.x * 256 + threadIdx.x;
    if (c < cols)
        out[(size_t)b * outBatch + (size_t)r * ldo + c] =
            (__bf16)in[(size_t)b * inBatch + (size_t)r * cols + c];
}

// ---------------- f32 -> bf16 tiled transpose into padded layout ----------------
__global__ __launch_bounds__(256) void transpose_f32_bf16(const float* __restrict__ in,
                                                          __bf16* __restrict__ out,
                                                          int rows, int cols, int ldo,
                                                          size_t outBatch) {
    __shared__ float tile[32][33];
    int b = blockIdx.z;
    int colBase = blockIdx.x * 32;
    int rowBase = blockIdx.y * 32;
    int tx = threadIdx.x & 31;
    int ty = threadIdx.x >> 5;              // 0..7
    const float* src = in + (size_t)b * rows * cols;
    __bf16* dst = out + (size_t)b * outBatch;
#pragma unroll
    for (int i = 0; i < 4; ++i) {
        int r = rowBase + ty + i * 8;
        int c = colBase + tx;
        tile[ty + i * 8][tx] = (r < rows && c < cols) ? src[(size_t)r * cols + c] : 0.0f;
    }
    __syncthreads();
#pragma unroll
    for (int i = 0; i < 4; ++i) {
        int c = colBase + ty + i * 8;       // output row index (original col)
        int r = rowBase + tx;               // output col index (original row)
        if (c < cols && r < rows)
            dst[(size_t)c * ldo + r] = (__bf16)tile[tx][ty + i * 8];
    }
}

// ---------------- input projection: x = nf @ W_in  ([*,16] x [16,128]) ----------------
__global__ __launch_bounds__(128) void input_proj(const float* __restrict__ nf,
                                                  const float* __restrict__ Win,
                                                  __bf16* __restrict__ x) {
    int row = blockIdx.x;                   // 0 .. B*P*N-1
    int z = row / Nc;
    int n = row % Nc;
    int c = threadIdx.x;                    // 0 .. 127
    const float* f = nf + (size_t)row * INc;
    float acc = 0.0f;
#pragma unroll
    for (int k = 0; k < INc; ++k) acc = fmaf(f[k], Win[k * HIDc + c], acc);
    x[((size_t)z * NPAD + n) * HIDc + c] = (__bf16)acc;
}

// ---------------- generic bf16 WMMA GEMM (fully padded: no guards) ----------------
// C[z] (+relu) = A[z/aDiv] @ B[z]
// A: [Mpad, K] row-major (lda == K), B: [K, ncols], C: [Mpad, ncols]
// grid.x = Mpad/256, grid.y = ncols/128, grid.z = batch
// 256 threads = 8 waves; block tile 256x128 (32 rows/wave, 2 A-frags, 16 WMMA/step);
// K-step 32; double-buffered LDS.
__global__ __launch_bounds__(256) void gemm_bf16_wmma(const __bf16* __restrict__ A,
                                                      const __bf16* __restrict__ Bm,
                                                      __bf16* __restrict__ C,
                                                      int K, int ncols,
                                                      size_t aStride, int aDiv,
                                                      size_t bStride, size_t cStride,
                                                      int relu) {
    __shared__ __bf16 As[2][256][32];       // [m][k]
    __shared__ __bf16 Bs[2][128][40];       // [n][k], padded stride (80B)

    int z = blockIdx.z;
    A  += (size_t)(z / aDiv) * aStride;
    Bm += (size_t)z * bStride;
    C  += (size_t)z * cStride;

    int row0 = blockIdx.x * 256;
    int col0 = blockIdx.y * 128;

    int t     = threadIdx.x;
    int lane  = t & 31;
    int wave  = t >> 5;                     // 0..7 -> 32-row strip
    int khalf = lane >> 4;                  // 0/1
    int nlane = lane & 15;

    // staging assignments (branch-free, everything in-bounds by padding)
    // A: one full 32-elem row per thread (4 x uint4)
    int bk  = t >> 3;                       // B: 0..31 k
    int bn0 = (t & 7) * 16;                 // B: 16 contiguous cols

    v8f acc0[8], acc1[8];
#pragma unroll
    for (int j = 0; j < 8; ++j) { acc0[j] = (v8f)(0.0f); acc1[j] = (v8f)(0.0f); }

    auto stage = [&](int buf, int kb) {
        const __bf16* asrc = A + (size_t)(row0 + t) * K + kb;
#pragma unroll
        for (int q = 0; q < 4; ++q)
            *(uint4*)&As[buf][t][q * 8] = *(const uint4*)(asrc + q * 8);
        const __bf16* bsrc = Bm + (size_t)(kb + bk) * ncols + col0 + bn0;
        __attribute__((aligned(16))) __bf16 tmp[16];
        *(uint4*)&tmp[0] = *(const uint4*)bsrc;
        *(uint4*)&tmp[8] = *(const uint4*)(bsrc + 8);
#pragma unroll
        for (int i = 0; i < 16; ++i) Bs[buf][bn0 + i][bk] = tmp[i];
    };

    stage(0, 0);
    __syncthreads();

    int buf = 0;
    int m = wave * 32 + nlane;
    for (int kb = 0; kb < K; kb += 32) {
        int nkb = kb + 32;
        if (nkb < K) {
            stage(buf ^ 1, nkb);
            // prefetch two K-steps ahead into cache (global_prefetch_b8)
            if (nkb + 32 < K) {
                __builtin_prefetch(A + (size_t)(row0 + t) * K + nkb + 32, 0, 1);
                __builtin_prefetch(Bm + (size_t)(nkb + 32 + bk) * ncols + col0 + bn0, 0, 1);
            }
        }
        // ---- two A fragments (doc 16x32 layout), rows m and m+16 ----
        v8bf a00 = *(const v8bf*)&As[buf][m][khalf * 8];
        v8bf a01 = *(const v8bf*)&As[buf][m][16 + khalf * 8];
        v16bf af0 = __builtin_shufflevector(a00, a01, 0,1,2,3,4,5,6,7,8,9,10,11,12,13,14,15);
        v8bf a10 = *(const v8bf*)&As[buf][m + 16][khalf * 8];
        v8bf a11 = *(const v8bf*)&As[buf][m + 16][16 + khalf * 8];
        v16bf af1 = __builtin_shufflevector(a10, a11, 0,1,2,3,4,5,6,7,8,9,10,11,12,13,14,15);
        // ---- per column tile: one B fragment feeds both row strips ----
#pragma unroll
        for (int j = 0; j < 8; ++j) {
            int n = j * 16 + nlane;
            v8bf b0 = *(const v8bf*)&Bs[buf][n][khalf * 8];
            v8bf b1 = *(const v8bf*)&Bs[buf][n][16 + khalf * 8];
            v16bf bfv = __builtin_shufflevector(b0, b1, 0,1,2,3,4,5,6,7,8,9,10,11,12,13,14,15);
            acc0[j] = __builtin_amdgcn_wmma_f32_16x16x32_bf16(
                false, af0, false, bfv, (short)0, acc0[j], false, false);
            acc1[j] = __builtin_amdgcn_wmma_f32_16x16x32_bf16(
                false, af1, false, bfv, (short)0, acc1[j], false, false);
        }
        __syncthreads();
        buf ^= 1;
    }

    // ---- epilogue (no guards: C rows are padded) ----
    int m0 = row0 + wave * 32;
#pragma unroll
    for (int j = 0; j < 8; ++j) {
        int col = col0 + j * 16 + nlane;
#pragma unroll
        for (int r = 0; r < 8; ++r) {
            int row = m0 + r + 8 * khalf;
            float v0 = acc0[j][r];
            float v1 = acc1[j][r];
            if (relu) { v0 = fmaxf(v0, 0.0f); v1 = fmaxf(v1, 0.0f); }
            C[(size_t)row * ncols + col]        = (__bf16)v0;
            C[(size_t)(row + 16) * ncols + col] = (__bf16)v1;
        }
    }
}

// ---------------- decode: out = 10*tanh((h2 @ W_dec)/sqrt(HID)) ----------------
__global__ __launch_bounds__(256) void decode_out(const __bf16* __restrict__ h2,
                                                  const float* __restrict__ wdec,
                                                  float* __restrict__ out, int rows) {
    int r = blockIdx.x * 256 + threadIdx.x;
    if (r >= rows) return;
    int z = r / Nc;
    int n = r % Nc;
    const __bf16* hr = h2 + ((size_t)z * NPAD + n) * HIDc;
    float acc = 0.0f;
#pragma unroll 8
    for (int j = 0; j < HIDc; ++j) acc = fmaf((float)hr[j], wdec[j], acc);
    acc *= 0.08838834764831845f;            // 1/sqrt(128)
    out[r] = 10.0f * tanhf(acc);
}

// ---------------- orchestration ----------------
extern "C" void kernel_launch(void* const* d_in, const int* in_sizes, int n_in,
                              void* d_out, int out_size, void* d_ws, size_t ws_size,
                              hipStream_t stream) {
    const float* nf      = (const float*)d_in[0];   // [B,P,N,IN]
    const float* H       = (const float*)d_in[1];   // [B,N,E]
    const float* Win     = (const float*)d_in[2];   // [IN,HID]
    const float* edgeWs  = (const float*)d_in[3];   // [NL,HID,HID]
    const float* nodeWs  = (const float*)d_in[4];   // [NL,HID,HID]
    const float* W1      = (const float*)d_in[5];   // [HID,4*HID]
    const float* W2      = (const float*)d_in[6];   // [4*HID,HID]
    const float* Wdec    = (const float*)d_in[7];   // [HID,1]
    float* out = (float*)d_out;

    char* ws = (char*)d_ws;
    size_t cur = 0;
    auto alloc = [&](size_t elems) -> __bf16* {
        cur = (cur + 255) & ~(size_t)255;
        __bf16* p = (__bf16*)(ws + cur);
        cur += elems * sizeof(__bf16);
        return p;
    };

    const size_t BP = (size_t)Bc * Pc;
    __bf16* Hbf   = alloc((size_t)Bc * NPAD * NPAD);    // H   [B, Npad, Epad] bf16
    __bf16* HTbf  = alloc((size_t)Bc * NPAD * NPAD);    // H^T [B, Epad, Npad] bf16
    __bf16* xbf   = alloc(BP * NPAD * HIDc);            // node activations
    __bf16* ebf   = alloc(BP * NPAD * HIDc);            // edge activations
    __bf16* tmpbf = alloc(BP * NPAD * HIDc);            // raw GEMM output
    __bf16* h1bf  = alloc(BP * NPAD * HIDc * MULTc);    // MLP hidden
    __bf16* ewbf  = alloc((size_t)NLc * HIDc * HIDc);
    __bf16* nwbf  = alloc((size_t)NLc * HIDc * HIDc);
    __bf16* w1bf  = alloc((size_t)HIDc * MULTc * HIDc);
    __bf16* w2bf  = alloc((size_t)MULTc * HIDc * HIDc);

    // 0) zero the whole used workspace (establishes zero padding everywhere)
    {
        size_t count16 = ((cur + 255) & ~(size_t)255) / 16;
        zero_ws<<<dim3(8192), 256, 0, stream>>>((uint4*)ws, count16);
    }

    // 1) one-time bf16 conversions (into padded layouts)
    cvt_pad_bf16<<<dim3((Ec + 255) / 256, Nc, Bc), 256, 0, stream>>>(
        H, Hbf, Ec, NPAD, (size_t)Nc * Ec, (size_t)NPAD * NPAD);
    transpose_f32_bf16<<<dim3((Ec + 31) / 32, (Nc + 31) / 32, Bc), 256, 0, stream>>>(
        H, HTbf, Nc, Ec, NPAD, (size_t)NPAD * NPAD);
    {
        long cw = (long)NLc * HIDc * HIDc;
        cvt_f32_bf16<<<dim3((unsigned)((cw + 255) / 256)), 256, 0, stream>>>(edgeWs, ewbf, cw);
        cvt_f32_bf16<<<dim3((unsigned)((cw + 255) / 256)), 256, 0, stream>>>(nodeWs, nwbf, cw);
        long c1 = (long)HIDc * MULTc * HIDc;
        cvt_f32_bf16<<<dim3((unsigned)((c1 + 255) / 256)), 256, 0, stream>>>(W1, w1bf, c1);
        cvt_f32_bf16<<<dim3((unsigned)((c1 + 255) / 256)), 256, 0, stream>>>(W2, w2bf, c1);
    }

    // 2) input projection: x = nf @ W_in
    input_proj<<<dim3((unsigned)(BP * Nc)), 128, 0, stream>>>(nf, Win, xbf);

    const dim3 blk(256);
    const dim3 gBig(NPAD / 256, 1, (unsigned)BP);
    const size_t actStride = (size_t)NPAD * HIDc;

    // 3) hypergraph layers
    for (int l = 0; l < NLc; ++l) {
        // e_raw = H^T @ x : A=HT[b] (Epad x Npad), B=x[z] (Npad x 128)
        gemm_bf16_wmma<<<gBig, blk, 0, stream>>>(
            HTbf, xbf, tmpbf, NPAD, HIDc,
            (size_t)NPAD * NPAD, Pc, actStride, actStride, 0);
        // e = relu(e_raw @ edge_Ws[l])
        gemm_bf16_wmma<<<gBig, blk, 0, stream>>>(
            tmpbf, ewbf + (size_t)l * HIDc * HIDc, ebf, HIDc, HIDc,
            actStride, 1, 0, actStride, 1);
        // x_raw = H @ e : A=H[b] (Npad x Epad), B=e[z] (Epad x 128)
        gemm_bf16_wmma<<<gBig, blk, 0, stream>>>(
            Hbf, ebf, tmpbf, NPAD, HIDc,
            (size_t)NPAD * NPAD, Pc, actStride, actStride, 0);
        // x = relu(x_raw @ node_Ws[l])
        gemm_bf16_wmma<<<gBig, blk, 0, stream>>>(
            tmpbf, nwbf + (size_t)l * HIDc * HIDc, xbf, HIDc, HIDc,
            actStride, 1, 0, actStride, 1);
    }

    // 4) MLP: h1 = relu(x @ W1)  (ncols = 512)
    gemm_bf16_wmma<<<dim3(NPAD / 256, MULTc, (unsigned)BP), blk, 0, stream>>>(
        xbf, w1bf, h1bf, HIDc, MULTc * HIDc,
        actStride, 1, 0, (size_t)NPAD * MULTc * HIDc, 1);
    //      h2 = h1 @ W2  (K = 512)
    gemm_bf16_wmma<<<gBig, blk, 0, stream>>>(
        h1bf, w2bf, tmpbf, MULTc * HIDc, HIDc,
        (size_t)NPAD * MULTc * HIDc, 1, 0, actStride, 0);

    // 5) decode: out = 10*tanh((h2 @ W_dec)/sqrt(HID))
    int rows = (int)(BP * Nc);
    decode_out<<<dim3((rows + 255) / 256), 256, 0, stream>>>(tmpbf, Wdec, out, rows);
}
